// GPT_Network_5377299055190
// MI455X (gfx1250) — matrix operand order repaired and checked by reading, not compiled, not run on hardware
//
#include <hip/hip_runtime.h>
#include <hip/hip_bf16.h>

typedef __attribute__((ext_vector_type(16))) _Float16 v16h;
typedef __attribute__((ext_vector_type(8)))  _Float16 v8h;
typedef __attribute__((ext_vector_type(4)))  _Float16 v4h;
typedef __attribute__((ext_vector_type(8)))  float    v8f;
typedef __attribute__((ext_vector_type(4)))  float    v4f;

#define HIDN   1024
#define SEQL   1024
#define NBATCH 2
#define NHEAD  16
#define DHEAD  64
#define FFWN   4096
#define NVOCAB 32000
#define EMBD   128
#define MTOK   (NBATCH * SEQL)   // 2048 tokens

// ---------------------------------------------------------------------------
// Embedding gather (float4): Aemb[t, e..e+3] = W_emb[x_input[t], e..e+3]
// ---------------------------------------------------------------------------
__global__ void k_gather_emb(const int* __restrict__ xin,
                             const float* __restrict__ Wemb,
                             float* __restrict__ out) {
  int q = blockIdx.x * blockDim.x + threadIdx.x;     // quad index
  if (q >= (MTOK * EMBD) / 4) return;
  int base = q * 4;
  int t = base >> 7;            // / EMBD
  int e = base & (EMBD - 1);
  v4f v = *(const v4f*)(Wemb + (size_t)xin[t] * EMBD + e);
  *(v4f*)(out + base) = v;
}

// ---------------------------------------------------------------------------
// Generic strided-batched WMMA GEMM:  C = op( (A @ B) (+bias) ) * scale
//   fp32 in/out, f16 WMMA compute (V_WMMA_F32_16X16X32_F16, f32 accum).
//   Block tile 128x64 x BK=64; 8 waves, each wave 32x32 (2x2 16x16 frags).
//   Pipeline: register prefetch of next K-tile + LDS double buffering,
//   single barrier per K-step.
//   z batching: offset = (z/zH)*s?1 + (z%zH)*s?2   (per-head attention views)
//   bias1: per-column [N];  bias2: per-(seq,col) [SEQ,biasLd] (row % SEQ)
// ---------------------------------------------------------------------------
#define AS 72          // lsA row stride (halves): 144B = 9*16B (16B-aligned frags)
#define BS 72          // lsBt col stride (halves)
#define ATILE (128 * AS)
#define BTILE (64 * BS)

__global__ __launch_bounds__(256)
void k_gemm_wmma(const float* __restrict__ A, size_t lda, size_t sA1, size_t sA2,
                 const float* __restrict__ B, size_t ldb, size_t sB1, size_t sB2,
                 float* __restrict__ C, size_t ldc, size_t sC1, size_t sC2,
                 int K, int zH, int transB,
                 const float* __restrict__ bias1,
                 const float* __restrict__ bias2, int biasLd,
                 float scale, int doRelu)
{
  __shared__ alignas(16) _Float16 lsA [2 * ATILE];  // [buf][row][k]  128 x 64
  __shared__ alignas(16) _Float16 lsBt[2 * BTILE];  // [buf][col][k]   64 x 64

  const int tid  = threadIdx.x;
  const int lane = tid & 31;
  const int wave = tid >> 5;
  const int wm   = wave & 3;   // 4 row groups of 32 -> 128
  const int wn   = wave >> 2;  // 2 col groups of 32 -> 64

  const int z  = blockIdx.z;
  const int zb = z / zH, zh = z % zH;
  const float* Ap = A + (size_t)zb * sA1 + (size_t)zh * sA2;
  const float* Bp = B + (size_t)zb * sB1 + (size_t)zh * sB2;
  float*       Cp = C + (size_t)zb * sC1 + (size_t)zh * sC2;

  const int rowBase = blockIdx.y * 128;
  const int colBase = blockIdx.x * 64;

  v8f acc[2][2];
#pragma unroll
  for (int i = 0; i < 2; ++i)
#pragma unroll
    for (int j = 0; j < 2; ++j)
#pragma unroll
      for (int r = 0; r < 8; ++r) acc[i][j][r] = 0.f;

  v4f ra[8];   // register-staged A quad-chunks (128x64 tile / 256 thr / 4)
  v4f rb[4];   // register-staged B quad-chunks ( 64x64 tile / 256 thr / 4)

  // Issue all global loads for the K-tile at k0 (no waits between them).
  auto loadTile = [&](int k0) {
#pragma unroll
    for (int u = 0; u < 8; ++u) {
      const int e = tid * 4 + u * 1024;
      const int r = e >> 6, c = e & 63;
      ra[u] = *(const v4f*)(Ap + (size_t)(rowBase + r) * lda + (size_t)(k0 + c));
    }
    if (transB) {
#pragma unroll
      for (int u = 0; u < 4; ++u) {
        const int e = tid * 4 + u * 1024;
        const int rr = e & 63, cc = e >> 6;     // k fast
        rb[u] = *(const v4f*)(Bp + (size_t)(colBase + cc) * ldb + (size_t)(k0 + rr));
      }
    } else {
#pragma unroll
      for (int u = 0; u < 4; ++u) {
        const int e = tid * 4 + u * 1024;
        const int r = e >> 6, c = e & 63;       // n fast
        rb[u] = *(const v4f*)(Bp + (size_t)(k0 + r) * ldb + (size_t)(colBase + c));
      }
    }
  };

  // Convert register-staged tile to f16 and store into LDS buffer `buf`.
  auto storeTile = [&](int buf) {
    _Float16* sA = lsA  + buf * ATILE;
    _Float16* sB = lsBt + buf * BTILE;
#pragma unroll
    for (int u = 0; u < 8; ++u) {
      const int e = tid * 4 + u * 1024;
      const int r = e >> 6, c = e & 63;
      v4h h;
#pragma unroll
      for (int i = 0; i < 4; ++i) h[i] = (_Float16)ra[u][i];
      *(v4h*)(&sA[r * AS + c]) = h;
    }
    if (transB) {
#pragma unroll
      for (int u = 0; u < 4; ++u) {
        const int e = tid * 4 + u * 1024;
        const int rr = e & 63, cc = e >> 6;
        v4h h;
#pragma unroll
        for (int i = 0; i < 4; ++i) h[i] = (_Float16)rb[u][i];
        *(v4h*)(&sB[cc * BS + rr]) = h;
      }
    } else {
#pragma unroll
      for (int u = 0; u < 4; ++u) {
        const int e = tid * 4 + u * 1024;
        const int r = e >> 6, c = e & 63;
#pragma unroll
        for (int i = 0; i < 4; ++i) sB[(c + i) * BS + r] = (_Float16)rb[u][i];
      }
    }
  };

  loadTile(0);
  int buf = 0;
  for (int k0 = 0; k0 < K; k0 += 64) {
    storeTile(buf);
    __syncthreads();
    if (k0 + 64 < K) loadTile(k0 + 64);   // next tile in flight during WMMAs

    const _Float16* sA = lsA  + buf * ATILE;
    const _Float16* sB = lsBt + buf * BTILE;
#pragma unroll
    for (int kk = 0; kk < 64; kk += 32) {
      // B frags (32x16): col-major LDS -> elems 0-7 = K kb..kb+7, elems 8-15 = K kb+16..kb+23
      v16h bfr[2];
      {
        const int kb = ((lane < 16) ? 0 : 8) + kk;
        const int nl = lane & 15;
#pragma unroll
        for (int j = 0; j < 2; ++j) {
          const int col = wn * 32 + j * 16 + nl;
          v8h b0 = *(const v8h*)(&sB[col * BS + kb]);
          v8h b1 = *(const v8h*)(&sB[col * BS + kb + 16]);
#pragma unroll
          for (int i = 0; i < 8; ++i) { bfr[j][i] = b0[i]; bfr[j][8 + i] = b1[i]; }
        }
      }
      // A frags (16x32): elems 0-7 = K kb..kb+7, elems 8-15 = K kb+16..kb+23
      v16h afr[2];
      {
        const int kb = ((lane < 16) ? 0 : 8) + kk;
        const int rl = lane & 15;
#pragma unroll
        for (int i = 0; i < 2; ++i) {
          const int row = wm * 32 + i * 16 + rl;
          v8h a0 = *(const v8h*)(&sA[row * AS + kb]);
          v8h a1 = *(const v8h*)(&sA[row * AS + kb + 16]);
#pragma unroll
          for (int e = 0; e < 8; ++e) { afr[i][e] = a0[e]; afr[i][8 + e] = a1[e]; }
        }
      }
#pragma unroll
      for (int i = 0; i < 2; ++i)
#pragma unroll
        for (int j = 0; j < 2; ++j)
          acc[i][j] = __builtin_amdgcn_wmma_f32_16x16x32_f16(
              false, afr[i], false, bfr[j], (short)0, acc[i][j], false, false);
    }
    buf ^= 1;
    // NOTE: no trailing barrier needed -- double buffering guarantees the
    // next store targets the buffer all waves finished computing on before
    // the barrier they already passed.
  }

  // C/D layout: VGPR r, lanes 0-15 -> M=r, lanes 16-31 -> M=r+8; N = lane&15
  const int cn = lane & 15;
  const int rO = (lane >> 4) * 8;
#pragma unroll
  for (int i = 0; i < 2; ++i) {
#pragma unroll
    for (int j = 0; j < 2; ++j) {
#pragma unroll
      for (int r = 0; r < 8; ++r) {
        const int gr = rowBase + wm * 32 + i * 16 + rO + r;
        const int gc = colBase + wn * 32 + j * 16 + cn;
        float v = acc[i][j][r];
        if (bias1) v += bias1[gc];
        if (bias2) v += bias2[(size_t)(gr & (SEQL - 1)) * biasLd + gc];
        v *= scale;
        if (doRelu) v = fmaxf(v, 0.f);
        Cp[(size_t)gr * ldc + gc] = v;
      }
    }
  }
}

// ---------------------------------------------------------------------------
// Block reductions: wave32 shuffle reduce + one LDS exchange (8 partials)
// ---------------------------------------------------------------------------
__device__ __forceinline__ float blk_sum(float v, float* part, int tid) {
#pragma unroll
  for (int m = 16; m > 0; m >>= 1) v += __shfl_xor(v, m, 32);
  if ((tid & 31) == 0) part[tid >> 5] = v;
  __syncthreads();
  float s = part[0];
#pragma unroll
  for (int w = 1; w < 8; ++w) s += part[w];
  return s;
}
__device__ __forceinline__ float blk_max(float v, float* part, int tid) {
#pragma unroll
  for (int m = 16; m > 0; m >>= 1) v = fmaxf(v, __shfl_xor(v, m, 32));
  if ((tid & 31) == 0) part[tid >> 5] = v;
  __syncthreads();
  float s = part[0];
#pragma unroll
  for (int w = 1; w < 8; ++w) s = fmaxf(s, part[w]);
  return s;
}

// ---------------------------------------------------------------------------
// Fused residual + LayerNorm: out[t,:] = add[t or t%SEQ, :] + LN(x[t,:])*gs+gb
// One 256-thread block per token; each thread owns one float4 of the row.
// ---------------------------------------------------------------------------
__global__ __launch_bounds__(256)
void k_ln_add(const float* __restrict__ x,
              const float* __restrict__ addsrc, int addPos,
              const float* __restrict__ gb, const float* __restrict__ gs,
              float* __restrict__ out)
{
  __shared__ float partA[8];
  __shared__ float partB[8];
  const int t   = blockIdx.x;
  const int tid = threadIdx.x;
  const float* xr = x + (size_t)t * HIDN;
  const int i0 = tid * 4;                       // 256 threads x 4 = 1024

  v4f xv = *(const v4f*)(xr + i0);
  const float mean =
      blk_sum(xv[0] + xv[1] + xv[2] + xv[3], partA, tid) * (1.f / HIDN);

  float vs = 0.f;
#pragma unroll
  for (int i = 0; i < 4; ++i) { float d = xv[i] - mean; vs += d * d; }
  const float rstd = rsqrtf(blk_sum(vs, partB, tid) * (1.f / HIDN) + 1e-6f);

  const float* ar = addsrc + (addPos ? (size_t)(t & (SEQL - 1)) * HIDN
                                     : (size_t)t * HIDN);
  v4f av  = *(const v4f*)(ar + i0);
  v4f gbv = *(const v4f*)(gb + i0);
  v4f gsv = *(const v4f*)(gs + i0);
  v4f ov;
#pragma unroll
  for (int i = 0; i < 4; ++i)
    ov[i] = (xv[i] - mean) * rstd * gsv[i] + gbv[i] + av[i];
  *(v4f*)(out + (size_t)t * HIDN + i0) = ov;
}

// ---------------------------------------------------------------------------
// Causal softmax over one score row (length SEQL); mask adds -1e9 for k>q
// grid = (SEQL, B*H); each thread owns one float4 of the row.
// ---------------------------------------------------------------------------
__global__ __launch_bounds__(256)
void k_softmax_causal(float* __restrict__ scores)
{
  __shared__ float partA[8];
  __shared__ float partB[8];
  const int q   = blockIdx.x;
  const int z   = blockIdx.y;
  const int tid = threadIdx.x;
  float* row = scores + ((size_t)z * SEQL + q) * SEQL;
  const int i0 = tid * 4;

  v4f v = *(const v4f*)(row + i0);
#pragma unroll
  for (int i = 0; i < 4; ++i) v[i] += ((i0 + i) <= q ? 0.f : -1.0e9f);

  const float mx =
      blk_max(fmaxf(fmaxf(v[0], v[1]), fmaxf(v[2], v[3])), partA, tid);

  v4f e;
  float sum = 0.f;
#pragma unroll
  for (int i = 0; i < 4; ++i) { e[i] = __expf(v[i] - mx); sum += e[i]; }
  const float inv = 1.f / blk_sum(sum, partB, tid);
#pragma unroll
  for (int i = 0; i < 4; ++i) e[i] *= inv;
  *(v4f*)(row + i0) = e;
}

// ---------------------------------------------------------------------------
extern "C" void kernel_launch(void* const* d_in, const int* in_sizes, int n_in,
                              void* d_out, int out_size, void* d_ws, size_t ws_size,
                              hipStream_t stream)
{
  (void)in_sizes; (void)n_in; (void)out_size; (void)ws_size;

  const int*   x_input   = (const int*)  d_in[0];
  const float* W_emb     = (const float*)d_in[1];
  const float* W_dec_lin = (const float*)d_in[2];
  const float* p_decoder = (const float*)d_in[3];
  const float* pos_embed = (const float*)d_in[4];
  const float* b_sq      = (const float*)d_in[5];
  const float* b_sk      = (const float*)d_in[6];
  const float* p_sq      = (const float*)d_in[7];
  const float* p_sk      = (const float*)d_in[8];
  const float* p_sv      = (const float*)d_in[9];
  const float* p_sc      = (const float*)d_in[10];
  const float* p_ff1     = (const float*)d_in[11];
  const float* p_ff2     = (const float*)d_in[12];
  const float* b_ff1     = (const float*)d_in[13];
  const float* b_ff2     = (const float*)d_in[14];
  const float* i_bias    = (const float*)d_in[15];
  const float* i_scale   = (const float*)d_in[16];
  const float* b_bias_1  = (const float*)d_in[17];
  const float* b_scale_1 = (const float*)d_in[18];
  const float* b_bias_2  = (const float*)d_in[19];
  const float* b_scale_2 = (const float*)d_in[20];
  const float* o_bias    = (const float*)d_in[21];
  const float* o_scale   = (const float*)d_in[22];
  float* out = (float*)d_out;

  // Workspace carve-out (fp32 elements)
  float* wsf = (float*)d_ws;
  size_t off = 0;
  auto carve = [&](size_t n) { float* p = wsf + off; off += n; return p; };
  float* Aemb   = carve((size_t)MTOK * EMBD);
  float* decin  = carve((size_t)MTOK * HIDN);
  float* cur    = carve((size_t)MTOK * HIDN);
  float* qb     = carve((size_t)MTOK * HIDN);
  float* kb     = carve((size_t)MTOK * HIDN);
  float* vb     = carve((size_t)MTOK * HIDN);
  float* ctx    = carve((size_t)MTOK * HIDN);
  float* xm     = carve((size_t)MTOK * HIDN);   // also reused for dec_outputs
  float* xself  = carve((size_t)MTOK * HIDN);
  float* ffo    = carve((size_t)MTOK * HIDN);
  float* ffh    = carve((size_t)MTOK * FFWN);
  float* scores = carve((size_t)NBATCH * NHEAD * SEQL * SEQL);

  const dim3 blk(256);
  auto gemm = [&](const float* A, size_t lda, size_t sA1, size_t sA2,
                  const float* B, size_t ldb, size_t sB1, size_t sB2,
                  float* C, size_t ldc, size_t sC1, size_t sC2,
                  int M, int N, int K, int Z, int zH, int transB,
                  const float* bias1, const float* bias2,
                  float scale, int relu) {
    dim3 grid((unsigned)(N / 64), (unsigned)(M / 128), (unsigned)Z);
    k_gemm_wmma<<<grid, blk, 0, stream>>>(A, lda, sA1, sA2, B, ldb, sB1, sB2,
                                          C, ldc, sC1, sC2, K, zH, transB,
                                          bias1, bias2, HIDN, scale, relu);
  };

  const size_t LHH = (size_t)HIDN * HIDN;      // per-layer HxH weight stride
  const size_t LF1 = (size_t)HIDN * FFWN;
  const size_t LF2 = (size_t)FFWN * HIDN;
  const size_t LSH = (size_t)SEQL * HIDN;      // per-layer [S,H] stride
  const float qscale = 0.125f;                 // 1/sqrt(64)

  // dec_inputs = W_emb[x_input] @ W_dec_lin
  k_gather_emb<<<(MTOK * EMBD / 4 + 255) / 256, blk, 0, stream>>>(x_input, W_emb, Aemb);
  gemm(Aemb, EMBD, 0, 0, W_dec_lin, HIDN, 0, 0, decin, HIDN, 0, 0,
       MTOK, HIDN, EMBD, 1, 1, 0, nullptr, nullptr, 1.f, 0);

  const float* src = decin;
  for (int m = 0; m < 12; ++m) {
    // layer_in = pos_embed[m] + LN(src)
    k_ln_add<<<MTOK, blk, 0, stream>>>(src, pos_embed + (size_t)m * LSH, 1,
                                       i_bias + (size_t)m * HIDN,
                                       i_scale + (size_t)m * HIDN, cur);
    // q/k/v projections (q has 1/sqrt(d) folded in, q/k have positional bias)
    gemm(cur, HIDN, 0, 0, p_sq + m * LHH, HIDN, 0, 0, qb, HIDN, 0, 0,
         MTOK, HIDN, HIDN, 1, 1, 0, nullptr, b_sq + m * LSH, qscale, 0);
    gemm(cur, HIDN, 0, 0, p_sk + m * LHH, HIDN, 0, 0, kb, HIDN, 0, 0,
         MTOK, HIDN, HIDN, 1, 1, 0, nullptr, b_sk + m * LSH, 1.f, 0);
    gemm(cur, HIDN, 0, 0, p_sv + m * LHH, HIDN, 0, 0, vb, HIDN, 0, 0,
         MTOK, HIDN, HIDN, 1, 1, 0, nullptr, nullptr, 1.f, 0);
    // scores[b,h] = q[b,h] @ k[b,h]^T   (z = b*16+h, per-head strided views)
    gemm(qb, HIDN, LSH, DHEAD,
         kb, HIDN, LSH, DHEAD,
         scores, SEQL, (size_t)NHEAD * SEQL * SEQL, (size_t)SEQL * SEQL,
         SEQL, SEQL, DHEAD, NBATCH * NHEAD, NHEAD, 1, nullptr, nullptr, 1.f, 0);
    // causal softmax
    k_softmax_causal<<<dim3(SEQL, NBATCH * NHEAD), blk, 0, stream>>>(scores);
    // ctx[b,h] = alphas @ v[b,h]  -> written into [B,S,HID] layout
    gemm(scores, SEQL, (size_t)NHEAD * SEQL * SEQL, (size_t)SEQL * SEQL,
         vb, HIDN, LSH, DHEAD,
         ctx, HIDN, LSH, DHEAD,
         SEQL, DHEAD, SEQL, NBATCH * NHEAD, NHEAD, 0, nullptr, nullptr, 1.f, 0);
    // x_multi = ctx @ p_sc
    gemm(ctx, HIDN, 0, 0, p_sc + m * LHH, HIDN, 0, 0, xm, HIDN, 0, 0,
         MTOK, HIDN, HIDN, 1, 1, 0, nullptr, nullptr, 1.f, 0);
    // x_self = layer_in + LN(x_multi)
    k_ln_add<<<MTOK, blk, 0, stream>>>(xm, cur, 0,
                                       b_bias_1 + (size_t)m * HIDN,
                                       b_scale_1 + (size_t)m * HIDN, xself);
    // ffw = relu(x_self @ p_ff1 + b_ff1) @ p_ff2 + b_ff2
    gemm(xself, HIDN, 0, 0, p_ff1 + m * LF1, FFWN, 0, 0, ffh, FFWN, 0, 0,
         MTOK, FFWN, HIDN, 1, 1, 0, b_ff1 + (size_t)m * FFWN, nullptr, 1.f, 1);
    gemm(ffh, FFWN, 0, 0, p_ff2 + m * LF2, HIDN, 0, 0, ffo, HIDN, 0, 0,
         MTOK, HIDN, FFWN, 1, 1, 0, b_ff2 + (size_t)m * HIDN, nullptr, 1.f, 0);
    // layer_in = x_self + LN(ffw)
    k_ln_add<<<MTOK, blk, 0, stream>>>(ffo, xself, 0,
                                       b_bias_2 + (size_t)m * HIDN,
                                       b_scale_2 + (size_t)m * HIDN, cur);
    src = cur;
  }

  // dec_outputs = dec_inputs + LN(layer_in)
  k_ln_add<<<MTOK, blk, 0, stream>>>(cur, decin, 0, o_bias, o_scale, xm);
  // logits = dec_outputs @ p_decoder
  gemm(xm, HIDN, 0, 0, p_decoder, NVOCAB, 0, 0, out, NVOCAB, 0, 0,
       MTOK, NVOCAB, HIDN, 1, 1, 0, nullptr, nullptr, 1.f, 0);
}
